// geomorph_32968168964591
// MI455X (gfx1250) — matrix operand order repaired and checked by reading, not compiled, not run on hardware
//
#include <hip/hip_runtime.h>
#include <cstdint>

#define KGMM 10
#define GEPS 1e-15f

typedef __attribute__((ext_vector_type(16))) _Float16 v16h;
typedef __attribute__((ext_vector_type(8)))  float    v8f;

// ---------------------------------------------------------------------------
// Row-normalize feat_x / feat_y and concatenate into x[162, 256].
// grid = (162, 2), block = 128
// ---------------------------------------------------------------------------
__global__ void k_norm_concat(const float* __restrict__ fx, const float* __restrict__ fy,
                              float* __restrict__ out, int F) {
  __shared__ float red[128];
  int row  = blockIdx.x;
  int half = blockIdx.y;                 // 0 -> cols [0,F), 1 -> cols [F,2F)
  const float* src = half ? fy : fx;
  int t = threadIdx.x;
  float v = src[row * F + t];
  red[t] = v * v;
  __syncthreads();
  for (int s = F >> 1; s > 0; s >>= 1) {
    if (t < s) red[t] += red[t + s];
    __syncthreads();
  }
  float nrm = sqrtf(red[0]);
  out[row * (2 * F) + half * F + t] = v / nrm;
}

// ---------------------------------------------------------------------------
// C[M,N] = A[M,Kd] * B[Kd,N] (row-major f32), via v_wmma_f32_16x16x32_f16.
// block = 256 (8 waves); each wave owns a 16x32 output tile (one A fragment,
// two B fragments, two accumulators -> 2 wmma per 32-deep k-step).
// VGPR layouts as loaded here (wave32; contiguous k, not the ISA table order):
//   A 16x32 f16 : lane<16 -> row M=lane,    K in {0..15}
//                 lane>=16 -> row M=lane-16, K in {16..31}
//   B 32x16 f16 : element j -> row K = 16*(lane>=16) + j, col N = lane&15
//   C/D 16x16 f32: vgpr r -> row M = 8*(lane>=16) + r,    col N = lane&15
// Requirements met by all convs here: Kd % 32 == 0, N % 32 == 0.
// ---------------------------------------------------------------------------
__global__ void k_gemm_f16wmma(const float* __restrict__ A, const float* __restrict__ Bm,
                               float* __restrict__ C, int M, int N, int Kd) {
  int lane = threadIdx.x & 31;
  int wave = threadIdx.x >> 5;
  int n0 = blockIdx.x * 256 + wave * 32;
  if (n0 >= N) return;                       // whole-wave early out only
  int m0    = blockIdx.y * 16;
  int r16   = lane & 15;
  int khalf = lane >> 4;

  int ra = m0 + r16;
  if (ra > M - 1) ra = M - 1;                // clamp pad rows (results discarded)

  const float* Abase = A  + (size_t)ra * Kd + khalf * 16;
  const float* B0    = Bm + (size_t)(khalf * 16) * N + (n0 + r16);
  const float* B1    = B0 + 16;

  v8f acc0 = {0.f, 0.f, 0.f, 0.f, 0.f, 0.f, 0.f, 0.f};
  v8f acc1 = {0.f, 0.f, 0.f, 0.f, 0.f, 0.f, 0.f, 0.f};

  for (int k0 = 0; k0 < Kd; k0 += 32) {
    v16h a, b0, b1;
    const float* Ap = Abase + k0;
#pragma unroll
    for (int j = 0; j < 8; ++j) {
      a[j]     = (_Float16)Ap[j];
      a[j + 8] = (_Float16)Ap[j + 8];
    }
    const float* Bp0 = B0 + (size_t)k0 * N;
    const float* Bp1 = B1 + (size_t)k0 * N;
#pragma unroll
    for (int j = 0; j < 16; ++j) {
      b0[j] = (_Float16)Bp0[(size_t)j * N];
      b1[j] = (_Float16)Bp1[(size_t)j * N];
    }
    acc0 = __builtin_amdgcn_wmma_f32_16x16x32_f16(false, a, false, b0,
                                                  (short)0, acc0, false, false);
    acc1 = __builtin_amdgcn_wmma_f32_16x16x32_f16(false, a, false, b1,
                                                  (short)0, acc1, false, false);
  }

  int    mbase = m0 + khalf * 8;
  float* Crow  = C + (size_t)mbase * N + (n0 + r16);
  if (m0 + 16 <= M) {
    // fast path: entire 16-row tile in range -> unconditional store stream
#pragma unroll
    for (int r = 0; r < 8; ++r) {
      Crow[(size_t)r * N]      = acc0[r];
      Crow[(size_t)r * N + 16] = acc1[r];
    }
  } else {
#pragma unroll
    for (int r = 0; r < 8; ++r) {
      if (mbase + r < M) {
        Crow[(size_t)r * N]      = acc0[r];
        Crow[(size_t)r * N + 16] = acc1[r];
      }
    }
  }
}

// ---------------------------------------------------------------------------
// Per-node edge message + mean aggregation + root/bias/residual/ReLU epilogue.
// dst == repeat(arange(n), 6): node i owns edges 6i..6i+5, cnt == 6 always.
// grid = n, block = cout (64 or 128).
// ---------------------------------------------------------------------------
__global__ void k_msg(const float* __restrict__ xg, const float* __restrict__ xroot,
                      const float* __restrict__ bias,
                      const int*   __restrict__ src, const float* __restrict__ ps,
                      const float* __restrict__ mu,  const float* __restrict__ sigma,
                      const float* __restrict__ addbuf, float* __restrict__ out,
                      int cout, int relu) {
  __shared__ float smu[2 * KGMM];
  __shared__ float sinv[2 * KGMM];
  int i = blockIdx.x;
  int o = threadIdx.x;
  if (o < 2 * KGMM) {
    smu[o] = mu[o];
    float s = sigma[o];
    sinv[o] = -0.5f / (GEPS + s * s);
  }
  __syncthreads();

  const int Kc = KGMM * cout;
  float acc = 0.f;
#pragma unroll 1
  for (int j = 0; j < 6; ++j) {
    int e = i * 6 + j;
    int s = src[e];
    float p0 = ps[2 * e], p1 = ps[2 * e + 1];
    const float* xp = xg + (size_t)s * Kc + o;
    float m = 0.f;
#pragma unroll
    for (int k = 0; k < KGMM; ++k) {
      float d0 = p0 - smu[2 * k];
      float d1 = p1 - smu[2 * k + 1];
      float wk = __expf(d0 * d0 * sinv[2 * k] + d1 * d1 * sinv[2 * k + 1]);
      m = fmaf(wk, xp[(size_t)k * cout], m);
    }
    acc += m;
  }
  float r = acc * (1.f / 6.f) + xroot[(size_t)i * cout + o] + bias[o];
  if (addbuf) r += addbuf[(size_t)i * cout + o];
  if (relu) r = fmaxf(r, 0.f);
  out[(size_t)i * cout + o] = r;
}

// ---------------------------------------------------------------------------
// hex upsample: rows [0,prev) copied, rows [prev,ntot) = mean of two parents.
// ---------------------------------------------------------------------------
__global__ void k_upsample(const float* __restrict__ xin, const int* __restrict__ up,
                           float* __restrict__ xout, int prev, int ntot, int F) {
  int idx = blockIdx.x * blockDim.x + threadIdx.x;
  int total = ntot * F;
  if (idx >= total) return;
  int r = idx / F, c = idx - r * F;
  float v;
  if (r < prev) {
    v = xin[(size_t)r * F + c];
  } else {
    int a = up[(r - prev) * 2 + 0];
    int b = up[(r - prev) * 2 + 1];
    v = 0.5f * (xin[(size_t)a * F + c] + xin[(size_t)b * F + c]);
  }
  xout[idx] = v;
}

// ---------------------------------------------------------------------------
// Faithful hex pool: x[hex[:num]] -> [num,7,F] row-major, reinterpret as
// [num,F,7], mean over last axis. out[i,f] = mean_t flat[i, f*7+t].
// ---------------------------------------------------------------------------
__global__ void k_hexpool(const float* __restrict__ xin, const int* __restrict__ hexa,
                          float* __restrict__ xout, int num, int F) {
  int idx = blockIdx.x * blockDim.x + threadIdx.x;
  if (idx >= num * F) return;
  int i = idx / F, f = idx - i * F;
  float s = 0.f;
#pragma unroll
  for (int t = 0; t < 7; ++t) {
    int q = f * 7 + t;
    int j = q / F, c = q - j * F;
    s += xin[(size_t)hexa[i * 7 + j] * F + c];
  }
  xout[idx] = s * (1.f / 7.f);
}

// ---------------------------------------------------------------------------
// Row softmax, F == 64, block = 64.
// ---------------------------------------------------------------------------
__global__ void k_softmax(const float* __restrict__ xin, float* __restrict__ out, int F) {
  __shared__ float red[64];
  int i = blockIdx.x, t = threadIdx.x;
  float v = xin[(size_t)i * F + t];
  red[t] = v; __syncthreads();
  for (int s = F >> 1; s > 0; s >>= 1) { if (t < s) red[t] = fmaxf(red[t], red[t + s]); __syncthreads(); }
  float mx = red[0]; __syncthreads();
  float e = __expf(v - mx);
  red[t] = e; __syncthreads();
  for (int s = F >> 1; s > 0; s >>= 1) { if (t < s) red[t] += red[t + s]; __syncthreads(); }
  out[(size_t)i * F + t] = e / red[0];
}

// ---------------------------------------------------------------------------
// Host-side orchestration
// ---------------------------------------------------------------------------
struct ConvP { const float *g, *mu, *sigma, *root, *bias; };

static void run_conv(const float* x, int n, int cin, int cout, const ConvP& p,
                     const int* src, const float* ps, float* xg, float* xroot,
                     const float* addbuf, float* out, int relu, hipStream_t stream) {
  const int Kc = KGMM * cout;
  dim3 blk(256);
  dim3 g1((Kc + 255) / 256, (n + 15) / 16);
  k_gemm_f16wmma<<<g1, blk, 0, stream>>>(x, p.g, xg, n, Kc, cin);
  dim3 g2((cout + 255) / 256, (n + 15) / 16);
  k_gemm_f16wmma<<<g2, blk, 0, stream>>>(x, p.root, xroot, n, cout, cin);
  k_msg<<<dim3(n), dim3(cout), 0, stream>>>(xg, xroot, p.bias, src, ps, p.mu, p.sigma,
                                            addbuf, out, cout, relu);
}

extern "C" void kernel_launch(void* const* d_in, const int* in_sizes, int n_in,
                              void* d_out, int out_size, void* d_ws, size_t ws_size,
                              hipStream_t stream) {
  (void)in_sizes; (void)n_in; (void)out_size; (void)ws_size;
  const float* feat_x = (const float*)d_in[0];
  const float* feat_y = (const float*)d_in[1];

  auto conv = [&](int base) {
    ConvP p;
    p.g     = (const float*)d_in[base + 0];
    p.mu    = (const float*)d_in[base + 1];
    p.sigma = (const float*)d_in[base + 2];
    p.root  = (const float*)d_in[base + 3];
    p.bias  = (const float*)d_in[base + 4];
    return p;
  };
  // params flattened depth-first in insertion order
  ConvP r1c1 = conv(2),  r1c2 = conv(7),  r1sc = conv(12);
  ConvP r2c1 = conv(17), r2c2 = conv(22);
  ConvP r3c1 = conv(27), r3c2 = conv(32), r3sc = conv(37);
  ConvP r4c1 = conv(42), r4c2 = conv(47);
  ConvP r5c1 = conv(52), r5c2 = conv(57);

  // mesh (ei row 0 == src)
  const int*   ei162    = (const int*)  d_in[62];
  const float* ps162    = (const float*)d_in[63];
  const int*   ei642    = (const int*)  d_in[64];
  const float* ps642    = (const float*)d_in[65];
  const int*   ei2562   = (const int*)  d_in[66];
  const float* ps2562   = (const float*)d_in[67];
  const int*   ei10242  = (const int*)  d_in[68];
  const float* ps10242  = (const float*)d_in[69];
  const int*   ei40962  = (const int*)  d_in[70];
  const float* ps40962  = (const float*)d_in[71];
  const int*   up642    = (const int*)  d_in[72];
  const int*   up2562   = (const int*)  d_in[73];
  const int*   up10242  = (const int*)  d_in[74];
  const int*   up40962  = (const int*)  d_in[75];
  const int*   hex40962 = (const int*)  d_in[76];
  const int*   hex10242 = (const int*)  d_in[77];

  // workspace layout (floats); ~148 MB total, every buffer written before read
  const size_t PAD  = 4096;
  const size_t NACT = (size_t)40962 * 64 + PAD;       // ping/pong activations (162 lvl uses 256ch: 41472 floats, fits)
  float* A  = (float*)d_ws;
  float* B  = A  + NACT;
  float* H  = B  + NACT;                              // conv1 output
  float* XR = H  + NACT;                              // x @ root
  float* XS = XR + NACT;                              // shortcut result (max 2562*64)
  float* XG = XS + (262144 + PAD);                    // x @ g table: max 40962*640

  // 1) normalize + concat -> A[162,256]
  k_norm_concat<<<dim3(162, 2), dim3(128), 0, stream>>>(feat_x, feat_y, A, 128);

  // res1 @162 (256->128, shortcut)
  run_conv(A, 162, 256, 128, r1sc, ei162, ps162, XG, XR, nullptr, XS, 0, stream);
  run_conv(A, 162, 256, 128, r1c1, ei162, ps162, XG, XR, nullptr, H,  1, stream);
  run_conv(H, 162, 128, 128, r1c2, ei162, ps162, XG, XR, XS,      B,  1, stream);

  // upsample 162 -> 642 (F=128): B -> A
  { int tot = 642 * 128; k_upsample<<<(tot + 255) / 256, 256, 0, stream>>>(B, up642, A, 162, 642, 128); }

  // res2 @642 (128->128, identity residual)
  run_conv(A, 642, 128, 128, r2c1, ei642, ps642, XG, XR, nullptr, H, 1, stream);
  run_conv(H, 642, 128, 128, r2c2, ei642, ps642, XG, XR, A,       B, 1, stream);

  // upsample 642 -> 2562 (F=128): B -> A
  { int tot = 2562 * 128; k_upsample<<<(tot + 255) / 256, 256, 0, stream>>>(B, up2562, A, 642, 2562, 128); }

  // res3 @2562 (128->64, shortcut)
  run_conv(A, 2562, 128, 64, r3sc, ei2562, ps2562, XG, XR, nullptr, XS, 0, stream);
  run_conv(A, 2562, 128, 64, r3c1, ei2562, ps2562, XG, XR, nullptr, H,  1, stream);
  run_conv(H, 2562,  64, 64, r3c2, ei2562, ps2562, XG, XR, XS,      B,  1, stream);

  // upsample 2562 -> 10242 (F=64): B -> A
  { int tot = 10242 * 64; k_upsample<<<(tot + 255) / 256, 256, 0, stream>>>(B, up10242, A, 2562, 10242, 64); }

  // res4 @10242 (64->64, identity residual)
  run_conv(A, 10242, 64, 64, r4c1, ei10242, ps10242, XG, XR, nullptr, H, 1, stream);
  run_conv(H, 10242, 64, 64, r4c2, ei10242, ps10242, XG, XR, A,       B, 1, stream);

  // upsample 10242 -> 40962 (F=64): B -> A
  { int tot = 40962 * 64; k_upsample<<<(tot + 255) / 256, 256, 0, stream>>>(B, up40962, A, 10242, 40962, 64); }

  // res5 @40962 (64->64, identity residual)
  run_conv(A, 40962, 64, 64, r5c1, ei40962, ps40962, XG, XR, nullptr, H, 1, stream);
  run_conv(H, 40962, 64, 64, r5c2, ei40962, ps40962, XG, XR, A,       B, 1, stream);

  // hex pools: 40962 -> 10242 (B->A), 10242 -> 2562 (A->B)
  { int tot = 10242 * 64; k_hexpool<<<(tot + 255) / 256, 256, 0, stream>>>(B, hex40962, A, 10242, 64); }
  { int tot = 2562  * 64; k_hexpool<<<(tot + 255) / 256, 256, 0, stream>>>(A, hex10242, B, 2562,  64); }

  // row softmax -> d_out [2562, 64]
  k_softmax<<<dim3(2562), dim3(64), 0, stream>>>(B, (float*)d_out, 64);
}
